// Layer_44538810860058
// MI455X (gfx1250) — matrix operand order repaired and checked by reading, not compiled, not run on hardware
//
#include <hip/hip_runtime.h>

// ---------------------------------------------------------------------------
// CDNA5 (gfx1250, wave32) WMMA types: 16x16x32 bf16 -> f32 accumulate
// ---------------------------------------------------------------------------
typedef __attribute__((ext_vector_type(16))) __bf16 v16bf;
typedef __attribute__((ext_vector_type(8)))  float  v8f;
typedef __attribute__((ext_vector_type(4)))  int    v4i;

#define DEV static __device__ __forceinline__

constexpr int Bc   = 8;
constexpr int Sc   = 1024;
constexpr int Hc   = 1024;
constexpr int NHc  = 16;
constexpr int DHc  = 64;
constexpr int DFFc = 4096;
constexpr int Mc   = Bc * Sc;   // 8192 token rows

DEV unsigned short f2bf(float f) {
  // round-to-nearest-even f32 -> bf16
  unsigned u = __float_as_uint(f);
  u += 0x7FFFu + ((u >> 16) & 1u);
  return (unsigned short)(u >> 16);
}

union AF { v16bf v; unsigned u[8]; };

DEV v8f wmma_bf16(const AF& a, const AF& b, v8f c) {
  return __builtin_amdgcn_wmma_f32_16x16x32_bf16(
      /*neg_a=*/false, a.v, /*neg_b=*/false, b.v,
      /*c_mod=*/(short)0, c, /*reuse_a=*/false, /*reuse_b=*/false);
}

// ---------------------------------------------------------------------------
// Async global->LDS 16B copy (CDNA5 GLOBAL_LOAD_ASYNC_TO_LDS_B128, ASYNCcnt)
// with a synchronous fallback if the builtin is unavailable.
// Builtin prototype (from hipcc diagnostic): (v4i AS1* src, v4i AS3* dst,
// imm int offset, imm int cpol).
// ---------------------------------------------------------------------------
#if __has_builtin(__builtin_amdgcn_global_load_async_to_lds_b128)
#define HAVE_ASYNC_LDS 1
#else
#define HAVE_ASYNC_LDS 0
#endif

DEV void copy16_g2l(const unsigned short* g, unsigned short* l) {
#if HAVE_ASYNC_LDS
  typedef __attribute__((address_space(1))) v4i* gv4p;
  typedef __attribute__((address_space(3))) v4i* lv4p;
  __builtin_amdgcn_global_load_async_to_lds_b128(
      (gv4p)(v4i*)(unsigned short*)g, (lv4p)(v4i*)l, 0, 0);
#else
  *(uint4*)l = *(const uint4*)g;
#endif
}

DEV void async_wait() {
#if HAVE_ASYNC_LDS
  asm volatile("s_wait_asynccnt 0x0" ::: "memory");
#endif
}

// ---------------------------------------------------------------------------
// Pre-pass 1: x (f32 [M][H]) -> bf16 [M][H], vectorized
// ---------------------------------------------------------------------------
__global__ __launch_bounds__(256) void cvt_kernel(
    const float* __restrict__ in, unsigned short* __restrict__ out) {
  size_t i = ((size_t)blockIdx.x * 256 + threadIdx.x) * 4;
  float4 v = *(const float4*)&in[i];
  unsigned short o[4] = {f2bf(v.x), f2bf(v.y), f2bf(v.z), f2bf(v.w)};
  *(uint2*)&out[i] = *(const uint2*)o;
}

// ---------------------------------------------------------------------------
// Pre-pass 2: W (f32 [K][N]) -> W^T (bf16 [N][K]), 32x32 LDS-tiled transpose
// ---------------------------------------------------------------------------
__global__ __launch_bounds__(256) void wt_kernel(
    const float* __restrict__ w, unsigned short* __restrict__ wt, int K, int N) {
  __shared__ float t[32][33];
  const int k0 = blockIdx.y * 32, n0 = blockIdx.x * 32;
  const int tx = threadIdx.x, ty = threadIdx.y;   // 32 x 8
#pragma unroll
  for (int i = 0; i < 32; i += 8)
    t[ty + i][tx] = w[(size_t)(k0 + ty + i) * N + (n0 + tx)];
  __syncthreads();
#pragma unroll
  for (int i = 0; i < 32; i += 8)
    wt[(size_t)(n0 + ty + i) * K + (k0 + tx)] = f2bf(t[tx][ty + i]);
}

// ---------------------------------------------------------------------------
// GEMM: C[M,N] = act(A[M,K] * W[K,N] + bias[N]) with A bf16 [M][K],
// W pre-transposed bf16 [N][K]. Block 256 thr = 8 waves; block tile 128x128;
// wave tile 32x64; K-step 64; double-buffered LDS filled by async B128 DMA.
// LDS row stride 72 bf16 = 144B: 16B-aligned chunks AND conflict-free frags.
//   OMODE: 0 f32 row-major | 1 bf16 row-major | 2 bf16 [b,h,s,d] | 3 bf16 [b,h,d,s]
// ---------------------------------------------------------------------------
template <int OMODE, bool GELU>
__global__ __launch_bounds__(256) void gemm_kernel(
    const unsigned short* __restrict__ Ab, const unsigned short* __restrict__ Wt,
    const float* __restrict__ bias, void* __restrict__ Op, int K, int N) {
  __shared__ unsigned short As[2][128][72];
  __shared__ unsigned short Bs[2][128][72];

  const int tid  = threadIdx.x;
  const int m0   = blockIdx.y * 128;
  const int n0   = blockIdx.x * 128;
  const int wave = tid >> 5, lane = tid & 31;
  const int lrow = lane & 15, lhalf = lane >> 4;
  const int wm   = (wave >> 1) * 32;
  const int wn   = (wave & 1)  * 64;

  v8f acc[2][4];
#pragma unroll
  for (int i = 0; i < 2; ++i)
#pragma unroll
    for (int j = 0; j < 4; ++j) acc[i][j] = (v8f)0.0f;

  auto stage = [&](int buf, int k0) {
    // 128 rows x 64 bf16 = 1024 16B-chunks per tile; 4 chunks/thread/tile
#pragma unroll
    for (int i = 0; i < 4; ++i) {
      int idx = tid + i * 256;
      int r = idx >> 3, c = (idx & 7) * 8;
      copy16_g2l(&Ab[(size_t)(m0 + r) * K + k0 + c], &As[buf][r][c]);
      copy16_g2l(&Wt[(size_t)(n0 + r) * K + k0 + c], &Bs[buf][r][c]);
    }
  };

  const int nsteps = K >> 6;
  stage(0, 0);
  for (int s = 0; s < nsteps; ++s) {
    async_wait();
    __syncthreads();
    const int cur = s & 1;
    if (s + 1 < nsteps) stage(1 - cur, (s + 1) << 6);

    // A fragments: 2 m-tiles x 2 k-chunks (16x32 bf16 each)
    AF afr[2][2];
#pragma unroll
    for (int mt = 0; mt < 2; ++mt) {
      int row = wm + mt * 16 + lrow;
#pragma unroll
      for (int kc = 0; kc < 2; ++kc)
#pragma unroll
        for (int v = 0; v < 8; ++v) {
          int kk = (v < 4) ? (lhalf * 8 + 2 * v) : (16 + lhalf * 8 + 2 * (v - 4));
          afr[mt][kc].u[v] = *(const unsigned*)&As[cur][row][kc * 32 + kk];
        }
    }
    // B fragments built per n-tile to bound register pressure
#pragma unroll
    for (int nt = 0; nt < 4; ++nt) {
      int n = wn + nt * 16 + lrow;
      AF bfr[2];
#pragma unroll
      for (int kc = 0; kc < 2; ++kc)
#pragma unroll
        for (int v = 0; v < 8; ++v) {
          int kk = (v < 4) ? (lhalf * 8 + 2 * v) : (16 + lhalf * 8 + 2 * (v - 4));
          bfr[kc].u[v] = *(const unsigned*)&Bs[cur][n][kc * 32 + kk];
        }
#pragma unroll
      for (int mt = 0; mt < 2; ++mt) {
        acc[mt][nt] = wmma_bf16(afr[mt][0], bfr[0], acc[mt][nt]);
        acc[mt][nt] = wmma_bf16(afr[mt][1], bfr[1], acc[mt][nt]);
      }
    }
  }

  // Epilogue: bias (+ exact GELU) + layout-specific store
#pragma unroll
  for (int nt = 0; nt < 4; ++nt) {
    int col = n0 + wn + nt * 16 + lrow;
    float bv = bias[col];
#pragma unroll
    for (int mt = 0; mt < 2; ++mt) {
#pragma unroll
      for (int r = 0; r < 8; ++r) {
        int row = m0 + wm + mt * 16 + r + lhalf * 8;
        float val = acc[mt][nt][r] + bv;
        if (GELU) val = 0.5f * val * (1.0f + erff(val * 0.70710678118654752f));
        if (OMODE == 0) {
          ((float*)Op)[(size_t)row * N + col] = val;
        } else if (OMODE == 1) {
          ((unsigned short*)Op)[(size_t)row * N + col] = f2bf(val);
        } else {
          int b = row >> 10, s = row & (Sc - 1);
          int h = col >> 6,  d = col & (DHc - 1);
          size_t idx;
          if (OMODE == 2) idx = (((size_t)b * NHc + h) * Sc + s) * DHc + d;
          else            idx = (((size_t)b * NHc + h) * DHc + d) * Sc + s;
          ((unsigned short*)Op)[idx] = f2bf(val);
        }
      }
    }
  }
}

// ---------------------------------------------------------------------------
// Flash attention: one wave per (b, h, 16-query tile); 64-key blocks.
// Q,K in [bh][s][d] bf16; V transposed [bh][d][s] bf16. Out bf16 [M][H].
// ---------------------------------------------------------------------------
__global__ __launch_bounds__(32) void attn_kernel(
    const unsigned short* __restrict__ qb, const unsigned short* __restrict__ kb,
    const unsigned short* __restrict__ vtb, const int* __restrict__ amask,
    unsigned short* __restrict__ ab) {
  __shared__ unsigned short pls[16 * 64];   // P staging (C-layout -> A-layout)
  const int lane = threadIdx.x;
  const int lrow = lane & 15, lhalf = lane >> 4;
  const int q0 = blockIdx.x * 16;
  const int h  = blockIdx.y, b = blockIdx.z;
  const size_t bh = (size_t)b * NHc + h;
  const unsigned short* qp = qb + bh * Sc * DHc;
  const unsigned short* kp = kb + bh * Sc * DHc;
  const unsigned short* vp = vtb + bh * (size_t)DHc * Sc;

  AF qf[2];
#pragma unroll
  for (int c = 0; c < 2; ++c)
#pragma unroll
    for (int v = 0; v < 8; ++v) {
      int kk = (v < 4) ? (lhalf * 8 + 2 * v) : (16 + lhalf * 8 + 2 * (v - 4));
      qf[c].u[v] = *(const unsigned*)&qp[(size_t)(q0 + lrow) * DHc + c * 32 + kk];
    }

  float mrow[8], lsum[8];
  v8f accO[4];
#pragma unroll
  for (int r = 0; r < 8; ++r) { mrow[r] = -1e30f; lsum[r] = 0.0f; }
#pragma unroll
  for (int nt = 0; nt < 4; ++nt) accO[nt] = (v8f)0.0f;

  for (int j = 0; j < Sc / 64; ++j) {
    const int kbase = j * 64;
    v8f sacc[4];
#pragma unroll
    for (int nt = 0; nt < 4; ++nt) {
      int keyn = kbase + nt * 16 + lrow;
      AF bf0, bf1;
#pragma unroll
      for (int v = 0; v < 8; ++v) {
        int kk = (v < 4) ? (lhalf * 8 + 2 * v) : (16 + lhalf * 8 + 2 * (v - 4));
        bf0.u[v] = *(const unsigned*)&kp[(size_t)keyn * DHc + kk];
        bf1.u[v] = *(const unsigned*)&kp[(size_t)keyn * DHc + 32 + kk];
      }
      v8f t = (v8f)0.0f;
      t = wmma_bf16(qf[0], bf0, t);
      t = wmma_bf16(qf[1], bf1, t);
      float madd = (amask[b * Sc + keyn] == 1) ? -1e30f : 0.0f;
#pragma unroll
      for (int r = 0; r < 8; ++r) sacc[nt][r] = t[r] * 0.125f + madd;
    }
    // online softmax: butterflies over 16-lane halves == C-layout rows
#pragma unroll
    for (int r = 0; r < 8; ++r) {
      float t = fmaxf(fmaxf(sacc[0][r], sacc[1][r]), fmaxf(sacc[2][r], sacc[3][r]));
      t = fmaxf(t, __shfl_xor(t, 1));
      t = fmaxf(t, __shfl_xor(t, 2));
      t = fmaxf(t, __shfl_xor(t, 4));
      t = fmaxf(t, __shfl_xor(t, 8));
      float mnew = fmaxf(mrow[r], t);
      float corr = __expf(mrow[r] - mnew);
      mrow[r] = mnew;
      float rsum = 0.0f;
#pragma unroll
      for (int nt = 0; nt < 4; ++nt) {
        float p = __expf(sacc[nt][r] - mnew);
        sacc[nt][r] = p;
        rsum += p;
      }
      rsum += __shfl_xor(rsum, 1);
      rsum += __shfl_xor(rsum, 2);
      rsum += __shfl_xor(rsum, 4);
      rsum += __shfl_xor(rsum, 8);
      lsum[r] = lsum[r] * corr + rsum;
#pragma unroll
      for (int nt = 0; nt < 4; ++nt) accO[nt][r] = accO[nt][r] * corr;
    }
    // P: C-layout -> LDS -> A-layout bf16
#pragma unroll
    for (int nt = 0; nt < 4; ++nt)
#pragma unroll
      for (int r = 0; r < 8; ++r)
        pls[(r + lhalf * 8) * 64 + nt * 16 + lrow] = f2bf(sacc[nt][r]);
    __syncthreads();
    AF pf[2];
#pragma unroll
    for (int c = 0; c < 2; ++c)
#pragma unroll
      for (int v = 0; v < 8; ++v) {
        int kk = (v < 4) ? (lhalf * 8 + 2 * v) : (16 + lhalf * 8 + 2 * (v - 4));
        pf[c].u[v] = *(const unsigned*)&pls[lrow * 64 + c * 32 + kk];
      }
    __syncthreads();
#pragma unroll
    for (int nt = 0; nt < 4; ++nt) {
      int d = nt * 16 + lrow;
      AF vf0, vf1;
#pragma unroll
      for (int v = 0; v < 8; ++v) {
        int kk = (v < 4) ? (lhalf * 8 + 2 * v) : (16 + lhalf * 8 + 2 * (v - 4));
        vf0.u[v] = *(const unsigned*)&vp[(size_t)d * Sc + kbase + kk];
        vf1.u[v] = *(const unsigned*)&vp[(size_t)d * Sc + kbase + 32 + kk];
      }
      accO[nt] = wmma_bf16(pf[0], vf0, accO[nt]);
      accO[nt] = wmma_bf16(pf[1], vf1, accO[nt]);
    }
  }
#pragma unroll
  for (int nt = 0; nt < 4; ++nt)
#pragma unroll
    for (int r = 0; r < 8; ++r) {
      float val = accO[nt][r] / lsum[r];
      int srow = q0 + r + lhalf * 8;
      int col  = h * DHc + nt * 16 + lrow;
      ab[((size_t)b * Sc + srow) * Hc + col] = f2bf(val);
    }
}

// ---------------------------------------------------------------------------
// Residual + LayerNorm over H=1024: one 256-thread block per row.
// Safe in-place when a == outf. Optional bf16 secondary output.
// ---------------------------------------------------------------------------
__global__ __launch_bounds__(256) void ln_kernel(
    const float* __restrict__ a, const float* __restrict__ res,
    const float* __restrict__ g, const float* __restrict__ beta,
    float* __restrict__ outf, unsigned short* __restrict__ outb) {
  __shared__ float red[256];
  const int row = blockIdx.x, tid = threadIdx.x;
  const size_t base = (size_t)row * Hc;
  float v[4];
  float s1 = 0.0f;
#pragma unroll
  for (int i = 0; i < 4; ++i) {
    int c = tid + i * 256;
    v[i] = a[base + c] + res[base + c];
    s1 += v[i];
  }
  red[tid] = s1;
  __syncthreads();
  for (int off = 128; off > 0; off >>= 1) {
    if (tid < off) red[tid] += red[tid + off];
    __syncthreads();
  }
  float mean = red[0] * (1.0f / Hc);
  __syncthreads();
  float s2 = 0.0f;
#pragma unroll
  for (int i = 0; i < 4; ++i) { float d = v[i] - mean; s2 += d * d; }
  red[tid] = s2;
  __syncthreads();
  for (int off = 128; off > 0; off >>= 1) {
    if (tid < off) red[tid] += red[tid + off];
    __syncthreads();
  }
  float rs = rsqrtf(red[0] * (1.0f / Hc) + 1e-12f);
#pragma unroll
  for (int i = 0; i < 4; ++i) {
    int c = tid + i * 256;
    float o = (v[i] - mean) * rs * g[c] + beta[c];
    outf[base + c] = o;
    if (outb) outb[base + c] = f2bf(o);
  }
}

// ---------------------------------------------------------------------------
// Host-side orchestration
// ---------------------------------------------------------------------------
extern "C" void kernel_launch(void* const* d_in, const int* in_sizes, int n_in,
                              void* d_out, int out_size, void* d_ws, size_t ws_size,
                              hipStream_t stream) {
  (void)in_sizes; (void)n_in; (void)out_size; (void)ws_size;
  const float* x  = (const float*)d_in[0];
  const int*   am = (const int*)d_in[1];
  const float* Wq = (const float*)d_in[2];  const float* bq  = (const float*)d_in[3];
  const float* Wk = (const float*)d_in[4];  const float* bk  = (const float*)d_in[5];
  const float* Wv = (const float*)d_in[6];  const float* bv  = (const float*)d_in[7];
  const float* Wo = (const float*)d_in[8];  const float* bo  = (const float*)d_in[9];
  const float* g1 = (const float*)d_in[10]; const float* be1 = (const float*)d_in[11];
  const float* W1 = (const float*)d_in[12]; const float* b1  = (const float*)d_in[13];
  const float* W2 = (const float*)d_in[14]; const float* b2  = (const float*)d_in[15];
  const float* g2 = (const float*)d_in[16]; const float* be2 = (const float*)d_in[17];

  char* w = (char*)d_ws;
  const size_t MH2 = (size_t)Mc * Hc * 2;   // bf16 [M][H]
  const size_t MH4 = (size_t)Mc * Hc * 4;   // f32  [M][H]
  const size_t HH2 = (size_t)Hc * Hc * 2;   // bf16 1024x1024
  const size_t HF2 = (size_t)Hc * DFFc * 2; // bf16 1024x4096
  unsigned short* xb  = (unsigned short*)w; w += MH2;   // x bf16
  unsigned short* qb  = (unsigned short*)w; w += MH2;   // Q  [bh][s][d]
  unsigned short* kb  = (unsigned short*)w; w += MH2;   // K  [bh][s][d]
  unsigned short* vtb = (unsigned short*)w; w += MH2;   // V^T [bh][d][s]
  unsigned short* ab  = (unsigned short*)w; w += MH2;   // attn out [M][H]
  float*          of32= (float*)w;          w += MH4;   // O-proj out
  float*          y1f = (float*)w;          w += MH4;   // LN1 out f32
  unsigned short* y1b = (unsigned short*)w; w += MH2;   // LN1 out bf16
  unsigned short* m1b = (unsigned short*)w; w += (size_t)Mc * DFFc * 2; // FFN1 out
  unsigned short* wqT = (unsigned short*)w; w += HH2;
  unsigned short* wkT = (unsigned short*)w; w += HH2;
  unsigned short* wvT = (unsigned short*)w; w += HH2;
  unsigned short* woT = (unsigned short*)w; w += HH2;
  unsigned short* w1T = (unsigned short*)w; w += HF2;   // [DFF][H]
  unsigned short* w2T = (unsigned short*)w;             // [H][DFF]

  dim3 blk(256);
  dim3 tblk(32, 8);

  // Pre-pass: bf16 conversion + weight transposes (one-time per call)
  cvt_kernel<<<dim3(Mc * Hc / 1024), blk, 0, stream>>>(x, xb);
  wt_kernel<<<dim3(Hc / 32, Hc / 32),   tblk, 0, stream>>>(Wq, wqT, Hc, Hc);
  wt_kernel<<<dim3(Hc / 32, Hc / 32),   tblk, 0, stream>>>(Wk, wkT, Hc, Hc);
  wt_kernel<<<dim3(Hc / 32, Hc / 32),   tblk, 0, stream>>>(Wv, wvT, Hc, Hc);
  wt_kernel<<<dim3(Hc / 32, Hc / 32),   tblk, 0, stream>>>(Wo, woT, Hc, Hc);
  wt_kernel<<<dim3(DFFc / 32, Hc / 32), tblk, 0, stream>>>(W1, w1T, Hc, DFFc);
  wt_kernel<<<dim3(Hc / 32, DFFc / 32), tblk, 0, stream>>>(W2, w2T, DFFc, Hc);

  dim3 gHH(Hc / 128, Mc / 128);     // 8 x 64
  dim3 gHF(DFFc / 128, Mc / 128);   // 32 x 64

  gemm_kernel<2, false><<<gHH, blk, 0, stream>>>(xb, wqT, bq, qb, Hc, Hc);
  gemm_kernel<2, false><<<gHH, blk, 0, stream>>>(xb, wkT, bk, kb, Hc, Hc);
  gemm_kernel<3, false><<<gHH, blk, 0, stream>>>(xb, wvT, bv, vtb, Hc, Hc);
  attn_kernel<<<dim3(Sc / 16, NHc, Bc), dim3(32), 0, stream>>>(qb, kb, vtb, am, ab);
  gemm_kernel<0, false><<<gHH, blk, 0, stream>>>(ab, woT, bo, of32, Hc, Hc);
  ln_kernel<<<dim3(Mc), blk, 0, stream>>>(of32, x, g1, be1, y1f, y1b);
  gemm_kernel<1, true><<<gHF, blk, 0, stream>>>(y1b, w1T, b1, m1b, Hc, DFFc);
  gemm_kernel<0, false><<<gHH, blk, 0, stream>>>(m1b, w2T, b2, (float*)d_out, DFFc, Hc);
  ln_kernel<<<dim3(Mc), blk, 0, stream>>>((const float*)d_out, y1f, g2, be2,
                                          (float*)d_out, nullptr);
}
